// GraphConv_58823872086379
// MI455X (gfx1250) — compile-verified, matrix operand order checked by reading, not run on hardware
//
#include <hip/hip_runtime.h>

typedef __attribute__((ext_vector_type(16))) _Float16 v16h;
typedef __attribute__((ext_vector_type(8)))  _Float16 v8h;
typedef __attribute__((ext_vector_type(8)))  float    v8f;

#define NB   8
#define CH   64
#define NN   16384
#define AGG  9
#define OUTC 64
#define K1   585            // (CH+1)*AGG
#define K1P  608            // padded to 19 chunks of 32
#define MT   32             // nodes per block (2 subtiles of 16 per wave)

__device__ __forceinline__ float artanh_clip(float x) {
    x = fminf(fmaxf(x, -1.0f + 1e-6f), 1.0f - 1e-6f);
    return 0.5f * __logf((1.0f + x) / (1.0f - x));
}
__device__ __forceinline__ float lrelu(float x) { return fmaxf(x, 0.01f * x); }

// ---- Stage T: X [B,C,N] (fp32) -> VXh [B,N,C] (f16), coalesced read over N ----
__global__ __launch_bounds__(256) void k_transpose(const float* __restrict__ X,
                                                   _Float16* __restrict__ VXh) {
    size_t g = (size_t)blockIdx.x * 256 + threadIdx.x;   // over B*C*N
    size_t b = g >> 20;                                   // / (CH*NN)
    size_t r = g & ((1u << 20) - 1);
    size_t c = r >> 14;                                   // / NN
    size_t n = r & (NN - 1);
    VXh[((b * NN + n) << 6) + c] = (_Float16)X[g];
}

// ---- Stage A: per-node artanh(clip(dot(e1))), artanh(clip(dot(e2))) ----
__global__ __launch_bounds__(256) void k_nodedots(const float* __restrict__ X,
                                                  const float* __restrict__ e1,
                                                  const float* __restrict__ e2,
                                                  float2* __restrict__ t12) {
    __shared__ float sE1[CH], sE2[CH];
    if (threadIdx.x < CH) { sE1[threadIdx.x] = e1[threadIdx.x]; sE2[threadIdx.x] = e2[threadIdx.x]; }
    __syncthreads();
    size_t g = (size_t)blockIdx.x * 256 + threadIdx.x;    // over B*N
    size_t b = g >> 14;
    size_t n = g & (NN - 1);
    const float* xb = X + (b << 20) + n;                  // X[b][c][n], stride NN over c
    float d1 = 0.f, d2 = 0.f;
#pragma unroll 8
    for (int c = 0; c < CH; ++c) {
        float xv = xb[(size_t)c << 14];
        d1 += xv * sE1[c];
        d2 += xv * sE2[c];
    }
    t12[g] = make_float2(artanh_clip(d1), artanh_clip(d2));
}

// ---- Stage W: repack weights into col-major f16 B operands ----
__global__ __launch_bounds__(256) void k_packw(const float* __restrict__ gw,   // [64][65][9]
                                               const float* __restrict__ cw,   // [64][64]
                                               const float* __restrict__ ow,   // [64][128]
                                               _Float16* __restrict__ wB1,     // [64][608]
                                               _Float16* __restrict__ wB2,     // [64][64]
                                               _Float16* __restrict__ wB3) {   // [64][128]
    int g = blockIdx.x * 256 + threadIdx.x;
    if (g < OUTC * K1P) {
        int o = g / K1P, k = g - o * K1P;
        float v = 0.f;
        if (k < K1) {
            int a = k / (CH + 1), i = k - a * (CH + 1);
            v = gw[o * K1 + i * AGG + a];                 // w[o][i][a]
        }
        wB1[g] = (_Float16)v;
    } else if (g < OUTC * K1P + OUTC * CH) {
        int j = g - OUTC * K1P;
        wB2[j] = (_Float16)cw[j];
    } else if (g < OUTC * K1P + OUTC * CH + OUTC * 2 * OUTC) {
        int j = g - OUTC * K1P - OUTC * CH;
        wB3[j] = (_Float16)ow[j];
    }
}

// ---- Stage E: per-edge hyperEdge scalar ----
__global__ __launch_bounds__(256) void k_edges(const long long* __restrict__ ADJ,
                                               const float* __restrict__ DIS,
                                               const float2* __restrict__ t12,
                                               float* __restrict__ hyp) {
    size_t g = (size_t)blockIdx.x * 256 + threadIdx.x;    // over B*N*AGG
    size_t b = g / ((size_t)NN * AGG);
    size_t e = g - b * (size_t)NN * AGG;
    size_t n = e / AGG;
    long long idx = ADJ[g];
    float inv = 1.0f / (DIS[g] + 1e-10f);
    float t1 = t12[b * NN + (size_t)idx].x;               // neighbor dot (edge1W)
    float t2 = t12[b * NN + n].y;                         // center dot (edge2W)
    float s = (t1 + t2) * inv;
    hyp[g] = 1.0f / (1.0f + __expf(-s));
}

// ---- Main: gather + 3 fused WMMA GEMMs per 32-node tile ----
// Each wave owns a 16-wide outC slice and TWO 16-node M-subtiles: each B chunk
// is loaded once and consumed by two WMMAs (register-level B reuse, halves the
// dominant L2 weight traffic).
__global__ __launch_bounds__(128) void k_main(const _Float16* __restrict__ VXh,
                                              const long long* __restrict__ ADJ,
                                              const float* __restrict__ hyp,
                                              const _Float16* __restrict__ wB1,
                                              const _Float16* __restrict__ wB2,
                                              const _Float16* __restrict__ wB3,
                                              const float* __restrict__ gb,
                                              const float* __restrict__ ob,
                                              float* __restrict__ out) {
    __shared__ __align__(32) _Float16 A1[MT][K1P];        // 38,912 B: scaled gathered feats
    __shared__ __align__(32) _Float16 F[MT][2 * OUTC];    //  8,192 B: fused [g | c]

    const int tid = threadIdx.x;
    const int b   = blockIdx.x >> 9;                      // NN/32 = 512 tiles per batch
    const int n0  = (blockIdx.x & 511) << 5;

    // zero-pad A1 K-tail (585..607)
    for (int j = tid; j < MT * (K1P - K1); j += 128) {
        int r = j / (K1P - K1);
        A1[r][K1 + j % (K1P - K1)] = (_Float16)0.0f;
    }
    // gather MT*9 neighbor rows, scale by hyperEdge, append hyper as channel 64
    for (int r = tid; r < MT * AGG; r += 128) {
        int node = r / AGG, a = r - node * AGG;
        size_t e = ((size_t)b * NN + (n0 + node)) * AGG + a;
        long long idx = ADJ[e];
        float h = hyp[e];
        _Float16 hh = (_Float16)h;
        const v8h* src = (const v8h*)(VXh + (((size_t)b * NN + (size_t)idx) << 6));
        _Float16* dst = &A1[node][a * (CH + 1)];
#pragma unroll
        for (int j = 0; j < 8; ++j) {
            v8h v = src[j];
#pragma unroll
            for (int t = 0; t < 8; ++t) dst[8 * j + t] = v[t] * hh;
        }
        dst[CH] = hh;
    }
    __syncthreads();

    const int wave = tid >> 5, lane = tid & 31;
    const int l15  = lane & 15;
    const int hiK  = (lane >> 4) << 4;                    // K half-select: 0 or 16
    const int mOff = (lane >> 4) << 3;                    // row half-select: 0 or 8
    const int col  = (wave << 4) + l15;                   // output channel

    // GEMM1: g = A1[32x608] x wB1[608x64]  (19 chunks, B reused for 2 subtiles)
    v8f acc0 = {}, acc1 = {};
#pragma unroll
    for (int kc = 0; kc < K1P / 32; ++kc) {
        v16h bm = *(const v16h*)(wB1 + (size_t)col * K1P + kc * 32 + hiK);
        v16h a0 = *(const v16h*)(&A1[l15][kc * 32 + hiK]);
        v16h a1 = *(const v16h*)(&A1[16 + l15][kc * 32 + hiK]);
        acc0 = __builtin_amdgcn_wmma_f32_16x16x32_f16(false, a0, false, bm, (short)0, acc0, false, false);
        acc1 = __builtin_amdgcn_wmma_f32_16x16x32_f16(false, a1, false, bm, (short)0, acc1, false, false);
    }
    {
        float bias = gb[col];
#pragma unroll
        for (int r = 0; r < 8; ++r) {
            F[r + mOff][col]      = (_Float16)lrelu(acc0[r] + bias);
            F[16 + r + mOff][col] = (_Float16)lrelu(acc1[r] + bias);
        }
    }

    // GEMM2: c = VXh tile [32x64] x wB2[64x64]  (A fed straight from global/L2)
    v8f acc2a = {}, acc2b = {};
    const _Float16* vxTile = VXh + (((size_t)b * NN + n0) << 6);
#pragma unroll
    for (int kc = 0; kc < CH / 32; ++kc) {
        v16h bm = *(const v16h*)(wB2 + (size_t)col * CH + kc * 32 + hiK);
        v16h a0 = *(const v16h*)(vxTile + ((size_t)l15 << 6)        + kc * 32 + hiK);
        v16h a1 = *(const v16h*)(vxTile + ((size_t)(16 + l15) << 6) + kc * 32 + hiK);
        acc2a = __builtin_amdgcn_wmma_f32_16x16x32_f16(false, a0, false, bm, (short)0, acc2a, false, false);
        acc2b = __builtin_amdgcn_wmma_f32_16x16x32_f16(false, a1, false, bm, (short)0, acc2b, false, false);
    }
#pragma unroll
    for (int r = 0; r < 8; ++r) {
        F[r + mOff][OUTC + col]      = (_Float16)lrelu(acc2a[r]);
        F[16 + r + mOff][OUTC + col] = (_Float16)lrelu(acc2b[r]);
    }
    __syncthreads();

    // GEMM3: out = F[32x128] x wB3[128x64]  (4 chunks, B reused for 2 subtiles)
    v8f acc3a = {}, acc3b = {};
#pragma unroll
    for (int kc = 0; kc < (2 * OUTC) / 32; ++kc) {
        v16h bm = *(const v16h*)(wB3 + (size_t)col * (2 * OUTC) + kc * 32 + hiK);
        v16h a0 = *(const v16h*)(&F[l15][kc * 32 + hiK]);
        v16h a1 = *(const v16h*)(&F[16 + l15][kc * 32 + hiK]);
        acc3a = __builtin_amdgcn_wmma_f32_16x16x32_f16(false, a0, false, bm, (short)0, acc3a, false, false);
        acc3b = __builtin_amdgcn_wmma_f32_16x16x32_f16(false, a1, false, bm, (short)0, acc3b, false, false);
    }
    {
        float bias = ob[col];
        float* dst0 = out + (((size_t)b * OUTC + col) << 14) + n0 + mOff;       // subtile 0
        float* dst1 = dst0 + 16;                                                 // subtile 1
        float4 v;
        v.x = lrelu(acc3a[0] + bias); v.y = lrelu(acc3a[1] + bias);
        v.z = lrelu(acc3a[2] + bias); v.w = lrelu(acc3a[3] + bias);
        *(float4*)dst0 = v;
        v.x = lrelu(acc3a[4] + bias); v.y = lrelu(acc3a[5] + bias);
        v.z = lrelu(acc3a[6] + bias); v.w = lrelu(acc3a[7] + bias);
        *(float4*)(dst0 + 4) = v;
        v.x = lrelu(acc3b[0] + bias); v.y = lrelu(acc3b[1] + bias);
        v.z = lrelu(acc3b[2] + bias); v.w = lrelu(acc3b[3] + bias);
        *(float4*)dst1 = v;
        v.x = lrelu(acc3b[4] + bias); v.y = lrelu(acc3b[5] + bias);
        v.z = lrelu(acc3b[6] + bias); v.w = lrelu(acc3b[7] + bias);
        *(float4*)(dst1 + 4) = v;
    }
}

extern "C" void kernel_launch(void* const* d_in, const int* in_sizes, int n_in,
                              void* d_out, int out_size, void* d_ws, size_t ws_size,
                              hipStream_t stream) {
    const float*     X   = (const float*)d_in[0];
    const long long* ADJ = (const long long*)d_in[1];
    const float*     DIS = (const float*)d_in[2];
    const float*     e1  = (const float*)d_in[3];
    const float*     e2  = (const float*)d_in[4];
    const float*     gw  = (const float*)d_in[5];
    const float*     gb  = (const float*)d_in[6];
    const float*     cw  = (const float*)d_in[7];
    const float*     ow  = (const float*)d_in[8];
    const float*     ob  = (const float*)d_in[9];
    float* out = (float*)d_out;

    char* ws = (char*)d_ws;
    _Float16* VXh = (_Float16*)(ws);                       // 16 MB: [B][N][64] f16
    float2*   t12 = (float2*)  (ws + 16777216);            // 1 MB: per-node artanh dots
    float*    hyp = (float*)   (ws + 17825792);            // 4.5 MB: per-edge sigmoid
    _Float16* wB1 = (_Float16*)(ws + 22544384);            // [64][608] f16
    _Float16* wB2 = (_Float16*)(ws + 22622208);            // [64][64]  f16
    _Float16* wB3 = (_Float16*)(ws + 22630400);            // [64][128] f16

    k_transpose<<<(NB * CH * NN) / 256, 256, 0, stream>>>(X, VXh);
    k_nodedots <<<(NB * NN) / 256, 256, 0, stream>>>(X, e1, e2, t12);
    k_packw    <<<(OUTC * K1P + OUTC * CH + OUTC * 2 * OUTC + 255) / 256, 256, 0, stream>>>(
                   gw, cw, ow, wB1, wB2, wB3);
    k_edges    <<<(NB * NN * AGG) / 256, 256, 0, stream>>>(ADJ, DIS, t12, hyp);
    k_main     <<<NB * (NN / MT), 128, 0, stream>>>(VXh, ADJ, hyp, wB1, wB2, wB3, gb, ob, out);
}
